// MyDCNv2_33268816675419
// MI455X (gfx1250) — matrix operand order repaired and checked by reading, not compiled, not run on hardware
//
#include <hip/hip_runtime.h>

#define N_    8
#define CIN   256
#define COUT  256
#define H_    64
#define W_    64
#define KTAPS 9
#define KTOT  (CIN*KTAPS)      // 2304
#define NPIX  16               // output pixels per workgroup (one oh row segment)
#define CCH   64               // input channels per LDS chunk
#define KCH   (CCH*KTAPS)      // 576 K-rows per chunk
#define PITCHC (KCH + 8)       // 584 halves/pixel row -> 1168B: 16B aligned, bank-conflict-free
#define NPAIR (NPIX*KTAPS)     // 144 (pixel,tap) pairs
#define KSTEPS (KCH/32)        // 18 WMMA K-steps per chunk

typedef _Float16 half16 __attribute__((ext_vector_type(16)));
typedef float    float8 __attribute__((ext_vector_type(8)));
typedef float    float4v __attribute__((ext_vector_type(4)));
typedef int      int4v  __attribute__((ext_vector_type(4)));
typedef unsigned int uint32x4 __attribute__((ext_vector_type(4)));

union Frag16 { half16 h; uint32x4 q[2]; };

__global__ void dcnv2_cvt_weight(const float* __restrict__ w, _Float16* __restrict__ o) {
  int i = blockIdx.x * 256 + threadIdx.x;
  if (i < COUT * KTOT) o[i] = (_Float16)w[i];
}

__launch_bounds__(256)
__global__ void dcnv2_main(const float* __restrict__ x,
                           const float* __restrict__ offs,
                           const float* __restrict__ mask,
                           const _Float16* __restrict__ wgtH,
                           float* __restrict__ out) {
  __shared__ __align__(16) _Float16 colT[NPIX * PITCHC]; // 18688 B
  __shared__ __align__(16) int   cIdx[NPAIR][4];         // clamped plane indices
  __shared__ __align__(16) float cWgt[NPAIR][4];         // validity-zeroed weights

  const int bid = blockIdx.x;
  const int ow0 = (bid & 3) * NPIX;
  const int oh  = (bid >> 2) & 63;
  const int n   = bid >> 8;

  const int tid  = threadIdx.x;
  const int wave = tid >> 5;        // 0..7 -> 32 output channels each
  const int lane = tid & 31;
  const int grp  = lane >> 4;       // lane-group for WMMA K split
  const int l16  = lane & 15;

  // ---- once: bilinear corner indices + (mask*valid-folded) weights, 16px x 9taps ----
  for (int p = tid; p < NPAIR; p += 256) {
    const int px  = p / KTAPS;
    const int tap = p - px * KTAPS;
    const int ow  = ow0 + px;
    const int ky  = tap / 3, kx = tap - 3 * (tap / 3);
    const size_t ob = (((size_t)n * (2 * KTAPS) + 2 * tap) * H_ + oh) * W_ + ow;
    const float dy = offs[ob];
    const float dx = offs[ob + (size_t)H_ * W_];
    const float m  = mask[(((size_t)n * KTAPS + tap) * H_ + oh) * W_ + ow];
    const float py = (float)(oh + ky) + dy;   // padded-image coords (pad=1)
    const float qx = (float)(ow + kx) + dx;
    const float y0f = floorf(py), x0f = floorf(qx);
    const float ly = py - y0f, lx = qx - x0f;
    const int y0 = (int)y0f - 1;              // unpadded top-left corner
    const int x0 = (int)x0f - 1;
    const float wy0 = 1.f - ly, wx0 = 1.f - lx;
    const float ww[4] = { wy0 * wx0 * m, wy0 * lx * m, ly * wx0 * m, ly * lx * m };
    const int   yy[4] = { y0, y0, y0 + 1, y0 + 1 };
    const int   xx[4] = { x0, x0 + 1, x0, x0 + 1 };
#pragma unroll
    for (int j = 0; j < 4; ++j) {
      const bool ok = (yy[j] >= 0) & (yy[j] < H_) & (xx[j] >= 0) & (xx[j] < W_);
      const int iy = min(max(yy[j], 0), H_ - 1);
      const int ix = min(max(xx[j], 0), W_ - 1);
      cIdx[p][j] = iy * W_ + ix;              // always a valid address
      cWgt[p][j] = ok ? ww[j] : 0.f;          // zero-pad folded into weight
    }
  }
  __syncthreads();

  float8 acc0 = {};
  float8 acc1 = {};

  const _Float16* wrow0 = wgtH + (size_t)(wave * 32 + l16) * KTOT;
  const _Float16* wrow1 = wrow0 + (size_t)16 * KTOT;

  for (int cc = 0; cc < CIN / CCH; ++cc) {
    const int c0 = cc * CCH;
    // ---- phase A: branchless bilinear gather of a 64-channel chunk into LDS ----
#pragma unroll 2
    for (int i = tid; i < NPAIR * CCH; i += 256) {
      const int p  = i >> 6;        // pair index
      const int cl = i & 63;        // channel within chunk
      const int px  = p / KTAPS;
      const int tap = p - px * KTAPS;
      const int4v   id4 = *(const int4v*)cIdx[p];     // one ds_load_b128
      const float4v w4  = *(const float4v*)cWgt[p];   // one ds_load_b128
      const float* __restrict__ xp = x + ((size_t)n * CIN + (c0 + cl)) * (H_ * W_);
      float v = xp[id4.x] * w4.x;                     // 4 unconditional gathers
      v += xp[id4.y] * w4.y;
      v += xp[id4.z] * w4.z;
      v += xp[id4.w] * w4.w;
      colT[px * PITCHC + cl * KTAPS + tap] = (_Float16)v;
    }
    __syncthreads();

    // ---- phase B: WMMA GEMM over this K-chunk ----
#pragma unroll 2
    for (int s = 0; s < KSTEPS; ++s) {
      const int kkl = s * 32;             // K within chunk
      const int kkg = c0 * KTAPS + kkl;   // global K into weight rows
      __builtin_prefetch(wrow0 + kkg + 64, 0, 1);   // global_prefetch_b8
      Frag16 a0, a1, b;
      // A 16x32 f16 layout: grp0 holds K {0..7,16..23}, grp1 holds K {8..15,24..31}
      a0.q[0] = *(const uint32x4*)(wrow0 + kkg + 8 * grp);
      a0.q[1] = *(const uint32x4*)(wrow0 + kkg + 16 + 8 * grp);
      a1.q[0] = *(const uint32x4*)(wrow1 + kkg + 8 * grp);
      a1.q[1] = *(const uint32x4*)(wrow1 + kkg + 16 + 8 * grp);
      // B 32x16 f16 layout: column = lane%16, grp0 holds K {0..7,16..23}, grp1 holds K {8..15,24..31}
      const _Float16* bp = colT + l16 * PITCHC + kkl + 8 * grp;
      b.q[0] = *(const uint32x4*)(bp);
      b.q[1] = *(const uint32x4*)(bp + 16);
      acc0 = __builtin_amdgcn_wmma_f32_16x16x32_f16(false, a0.h, false, b.h,
                                                    (short)0, acc0, false, false);
      acc1 = __builtin_amdgcn_wmma_f32_16x16x32_f16(false, a1.h, false, b.h,
                                                    (short)0, acc1, false, false);
    }
    __syncthreads();
  }

  // ---- epilogue: D layout oc = base + r + 8*grp, column = lane%16 ----
  const size_t outBase = ((size_t)n * COUT) * (H_ * W_) + (size_t)oh * W_ + ow0 + l16;
  const int ocb0 = wave * 32 + 8 * grp;
#pragma unroll
  for (int r = 0; r < 8; ++r)
    out[outBase + (size_t)(ocb0 + r) * (H_ * W_)] = acc0[r];
  const int ocb1 = ocb0 + 16;
#pragma unroll
  for (int r = 0; r < 8; ++r)
    out[outBase + (size_t)(ocb1 + r) * (H_ * W_)] = acc1[r];
}

extern "C" void kernel_launch(void* const* d_in, const int* in_sizes, int n_in,
                              void* d_out, int out_size, void* d_ws, size_t ws_size,
                              hipStream_t stream) {
  const float* x    = (const float*)d_in[0];
  const float* offs = (const float*)d_in[1];
  const float* mask = (const float*)d_in[2];
  const float* w    = (const float*)d_in[3];
  _Float16* wgtH = (_Float16*)d_ws;   // 256*2304 halves = 1.125 MB scratch

  dcnv2_cvt_weight<<<(COUT * KTOT + 255) / 256, 256, 0, stream>>>(w, wgtH);
  dcnv2_main<<<N_ * H_ * (W_ / NPIX), 256, 0, stream>>>(x, offs, mask, wgtH,
                                                        (float*)d_out);
}
